// DeformConv2d_89154931130427
// MI455X (gfx1250) — matrix operand order repaired and checked by reading, not compiled, not run on hardware
//
#include <hip/hip_runtime.h>
#include <hip/hip_bf16.h>
#include <stdint.h>

#define B_    4
#define C_    64
#define CO_   64
#define H_    192
#define W_    192
#define KK_   9
#define PAD_  1
#define HW_   (H_ * W_)
#define CK_   (C_ * KK_)        // 576
#define PIX_TILE 32
#define THREADS  256

typedef __attribute__((ext_vector_type(16))) __bf16 v16bf;
typedef __attribute__((ext_vector_type(8)))  __bf16 v8bf;
typedef __attribute__((ext_vector_type(8)))  float  v8f;

// -------------------------------------------------------------------------
// Kernel 1: offset conv  p[b][m][hw] = bias[m] + sum_{c,3x3} w_off * x
// -------------------------------------------------------------------------
__global__ void __launch_bounds__(256)
offset_conv_kernel(const float* __restrict__ x,
                   const float* __restrict__ w_off,
                   const float* __restrict__ b_off,
                   float* __restrict__ p) {
    int gid = blockIdx.x * blockDim.x + threadIdx.x;
    if (gid >= B_ * 2 * KK_ * HW_) return;
    int hw = gid % HW_;
    int t  = gid / HW_;
    int m  = t % (2 * KK_);
    int b  = t / (2 * KK_);
    int h  = hw / W_, w = hw % W_;

    const float* xb = x + (size_t)b * C_ * HW_;
    const float* wr = w_off + (size_t)m * C_ * KK_;
    float acc = b_off[m];
    for (int c = 0; c < C_; ++c) {
        const float* xc = xb + (size_t)c * HW_;
        const float* wc = wr + c * KK_;
        #pragma unroll
        for (int ky = 0; ky < 3; ++ky) {
            int yy = h + ky - 1;
            if (yy < 0 || yy >= H_) continue;
            #pragma unroll
            for (int kx = 0; kx < 3; ++kx) {
                int xx = w + kx - 1;
                if (xx < 0 || xx >= W_) continue;
                acc += wc[ky * 3 + kx] * xc[yy * W_ + xx];
            }
        }
    }
    p[gid] = acc;   // gid enumeration == ((b*18 + m)*HW + hw)
}

// -------------------------------------------------------------------------
// Kernel 2: w_deform fp32 (CO,C,3,3) -> bf16 row-major [CO][CK] (same flat order)
// -------------------------------------------------------------------------
__global__ void __launch_bounds__(256)
wconvert_kernel(const float* __restrict__ wd, __bf16* __restrict__ wb) {
    int i = blockIdx.x * blockDim.x + threadIdx.x;
    if (i < CO_ * CK_) wb[i] = (__bf16)wd[i];
}

// -------------------------------------------------------------------------
// Kernel 3: fused bilinear sampling + WMMA GEMM
//   out[b][o][hw] = sum_{ck} wb[o][ck] * sampled[ck][hw]
// -------------------------------------------------------------------------
__global__ void __launch_bounds__(THREADS)
deform_wmma_kernel(const float* __restrict__ x,
                   const float* __restrict__ p,
                   const __bf16* __restrict__ wb,
                   float* __restrict__ out) {
    __shared__ __align__(32) __bf16 smp[PIX_TILE][CK_];   // 36 KB sampled tile
    __shared__ int   sIdx[4][KK_][PIX_TILE];              // 4.5 KB corner indices
    __shared__ float sWt [4][KK_][PIX_TILE];              // 4.5 KB corner weights

    const int tid    = threadIdx.x;
    const int blk    = blockIdx.x;
    const int b      = blk / (HW_ / PIX_TILE);
    const int hwbase = (blk % (HW_ / PIX_TILE)) * PIX_TILE;
    const float* xb  = x + (size_t)b * C_ * HW_;
    const float* pb  = p + (size_t)b * (2 * KK_) * HW_;

    // ---- Stage 1: bilinear indices/weights per (pixel, tap) ----
    for (int e = tid; e < PIX_TILE * KK_; e += THREADS) {
        int pp = e / KK_;
        int k  = e % KK_;
        int hw = hwbase + pp;
        int h  = hw / W_, w = hw % W_;
        // reference channel scramble: off[m] = p[2*(m%9) + m/9]
        int mdy = 2 * k, mdx = 2 * k + 1;
        int chy = 2 * (mdy % KK_) + (mdy / KK_);
        int chx = 2 * (mdx % KK_) + (mdx / KK_);
        float dy = pb[(size_t)chy * HW_ + hw];
        float dx = pb[(size_t)chx * HW_ + hw];
        float ys = (float)h + (float)(k / 3 - PAD_) + dy;
        float xs = (float)w + (float)(k % 3 - PAD_) + dx;
        float y0f = floorf(ys), x0f = floorf(xs);
        float wy1 = ys - y0f, wx1 = xs - x0f;
        float wy0 = 1.0f - wy1, wx0 = 1.0f - wx1;
        int y0 = (int)y0f, x0 = (int)x0f;
        #pragma unroll
        for (int corner = 0; corner < 4; ++corner) {
            int yi = y0 + (corner >> 1);
            int xi = x0 + (corner & 1);
            float wgt = ((corner >> 1) ? wy1 : wy0) * ((corner & 1) ? wx1 : wx0);
            bool valid = (yi >= 0) && (yi < H_) && (xi >= 0) && (xi < W_);
            int yc = min(max(yi, 0), H_ - 1);
            int xc = min(max(xi, 0), W_ - 1);
            sIdx[corner][k][pp] = yc * W_ + xc;
            sWt [corner][k][pp] = valid ? wgt : 0.0f;
        }
    }
    __syncthreads();

    // ---- Stage 2: gather + bilinear -> bf16 LDS tile [pixel][ck] ----
    // lane <-> pixel: 32 lanes of a wave gather 32 consecutive pixels of the
    // SAME (c, k) -> nearly-contiguous corner addresses (coalesced L2 reqs),
    // wave-uniform (c,k) address setup, conflict-free sIdx/sWt reads.
    for (int e = tid; e < CK_ * PIX_TILE; e += THREADS) {
        int pp = e & (PIX_TILE - 1);
        int ck = e >> 5;               // PIX_TILE == 32
        int c  = ck / KK_;
        int k  = ck % KK_;
        const float* xc = xb + (size_t)c * HW_;
        float v = sWt[0][k][pp] * xc[sIdx[0][k][pp]]
                + sWt[1][k][pp] * xc[sIdx[1][k][pp]]
                + sWt[2][k][pp] * xc[sIdx[2][k][pp]]
                + sWt[3][k][pp] * xc[sIdx[3][k][pp]];
        smp[pp][ck] = (__bf16)v;
    }
    __syncthreads();

    // ---- Stage 3: WMMA, 8 waves = 4 o-tiles x 2 pixel-tiles ----
    const int wid   = tid >> 5;
    const int lane  = tid & 31;
    const int ot    = wid >> 1;            // 0..3 : output-channel tile
    const int pt    = wid & 1;             // 0..1 : pixel tile
    const int lhalf = (lane < 16) ? 0 : 1; // K-half owned by this lane
    const int l15   = lane & 15;

    const __bf16* arow = wb + (size_t)(ot * 16 + l15) * CK_;
    const __bf16* brow = &smp[pt * 16 + l15][0];

    v8f acc = {};
    #pragma unroll
    for (int kc = 0; kc < CK_ / 32; ++kc) {           // 18 WMMA per wave
        // A (16x32 bf16): lanes 0-15 hold K 0-15; lanes 16-31 K 16-31
        int abase = kc * 32 + lhalf * 16;
        v8bf a0 = *(const v8bf*)(arow + abase);
        v8bf a1 = *(const v8bf*)(arow + abase + 8);
        v16bf a = __builtin_shufflevector(a0, a1, 0, 1, 2, 3, 4, 5, 6, 7,
                                                  8, 9, 10, 11, 12, 13, 14, 15);
        // B (32x16 bf16): lane = column, 16 K-contiguous values per lane
        v16bf bm = *(const v16bf*)(brow + kc * 32 + lhalf * 16);
        acc = __builtin_amdgcn_wmma_f32_16x16x32_bf16(
                  false, a, false, bm, (short)0, acc, false, false);
    }

    // ---- Stage 4: non-temporal store of D (never re-read; keep L2 for x/p/wb)
    float* ob = out + (size_t)b * CO_ * HW_;
    int pix = hwbase + pt * 16 + l15;
    #pragma unroll
    for (int v = 0; v < 8; ++v) {
        int o = ot * 16 + v + lhalf * 8;
        __builtin_nontemporal_store(acc[v], &ob[(size_t)o * HW_ + pix]);
    }
}

// -------------------------------------------------------------------------
extern "C" void kernel_launch(void* const* d_in, const int* in_sizes, int n_in,
                              void* d_out, int out_size, void* d_ws, size_t ws_size,
                              hipStream_t stream) {
    const float* x  = (const float*)d_in[0];   // (B, C, H, W)
    const float* wd = (const float*)d_in[1];   // (CO, C, 3, 3)
    const float* wo = (const float*)d_in[2];   // (18, C, 3, 3)
    const float* bo = (const float*)d_in[3];   // (18,)
    float* out = (float*)d_out;

    char* ws = (char*)d_ws;
    float* p = (float*)ws;                                    // 10.6 MB offsets
    size_t pBytes = (size_t)B_ * 2 * KK_ * HW_ * sizeof(float);
    __bf16* wb = (__bf16*)(ws + ((pBytes + 255) & ~(size_t)255)); // 72 KB bf16 W

    int total1 = B_ * 2 * KK_ * HW_;
    offset_conv_kernel<<<(total1 + 255) / 256, 256, 0, stream>>>(x, wo, bo, p);

    int total2 = CO_ * CK_;
    wconvert_kernel<<<(total2 + 255) / 256, 256, 0, stream>>>(wd, wb);

    int blocks = B_ * (HW_ / PIX_TILE);   // 4608
    deform_wmma_kernel<<<blocks, THREADS, 0, stream>>>(x, p, wb, out);
}
